// Ours_18829136625900
// MI455X (gfx1250) — matrix operand order repaired and checked by reading, not compiled, not run on hardware
//
#include <hip/hip_runtime.h>
#include <hip/hip_bf16.h>
#include <math.h>

typedef __attribute__((ext_vector_type(16))) _Float16 v16h;
typedef __attribute__((ext_vector_type(8)))  _Float16 v8h;
typedef __attribute__((ext_vector_type(4)))  _Float16 v4h;
typedef __attribute__((ext_vector_type(8)))  float    v8f;
typedef __attribute__((ext_vector_type(4)))  float    v4f;

#define T_TOK 10
#define DDIM  256
#define BATCH 64
#define NNODE 1024
#define MTOT  (T_TOK + NNODE)   /* 1034 */
#define KSEL  517               /* ceil(0.5 * 1034) */
#define INNER_TH (-0.847297861f)  /* log(0.3/0.7): sigmoid(z)>=0.3 <=> z>=th */
#define CROSS_TH (-2.197224577f)  /* log(0.1/0.9) */

__device__ __forceinline__ v16h cat8(v8h lo, v8h hi) {
    return __builtin_shufflevector(lo, hi, 0,1,2,3,4,5,6,7,8,9,10,11,12,13,14,15);
}

__device__ __forceinline__ v16h cvt16(v4f f0, v4f f1, v4f f2, v4f f3) {
    v16h r;
    #pragma unroll
    for (int i = 0; i < 4; ++i) {
        r[i]      = (_Float16)f0[i];
        r[4 + i]  = (_Float16)f1[i];
        r[8 + i]  = (_Float16)f2[i];
        r[12 + i] = (_Float16)f3[i];
    }
    return r;
}

// ---------------------------------------------------------------------------
// Kernel 1: cross-dot via WMMA. One wave (32 threads) per (batch, 16-col tile).
// crossdot[b, t(0..15), n] = tokens[t] . x[b,n]   (t>=T rows zero-padded)
// Last block computes tokdot[16x16] = tokens . tokens^T (cols padded).
// Tokens are staged zero-padded in LDS as f16 so every WMMA operand load is
// unconditional and wide (ds_load_b128 / global_load_b128).
// ---------------------------------------------------------------------------
__global__ __launch_bounds__(32) void cross_wmma_kernel(
        const float* __restrict__ x, const float* __restrict__ tokens,
        float* __restrict__ crossdot, float* __restrict__ tokdot)
{
    __shared__ __align__(16) _Float16 ltok[16 * DDIM];   // 8 KB, zero-padded rows 10..15

    const int lane = threadIdx.x;
    const int half = lane >> 4;     // 0: lanes 0-15, 1: lanes 16-31
    const int row  = lane & 15;     // A: M row; B: N column
    const int tile = blockIdx.x;
    const bool isTok = (tile == BATCH * (NNODE / 16));

    // --- stage padded f16 tokens in LDS (no divergent loads) ---
    {
        v4h* lt4 = (v4h*)ltok;
        const v4f* tk4 = (const v4f*)tokens;
        for (int i = lane; i < (T_TOK * DDIM) / 4; i += 32) {
            v4f t = tk4[i];
            v4h h;
            h[0] = (_Float16)t[0]; h[1] = (_Float16)t[1];
            h[2] = (_Float16)t[2]; h[3] = (_Float16)t[3];
            lt4[i] = h;
        }
        v4h z = {};
        for (int i = (T_TOK * DDIM) / 4 + lane; i < (16 * DDIM) / 4; i += 32)
            lt4[i] = z;
    }
    __syncthreads();

    const _Float16* arow = &ltok[row * DDIM];
    v8f acc = {};

    if (isTok) {
        // B[K][n] = padded tokens[n][K], from LDS
        for (int kc = 0; kc < DDIM; kc += 32) {
            // A 16x32 layout: lanes 0-15 K=kc..kc+15,
            //                 lanes16-31 K=kc+16..kc+31
            v16h a = cat8(*(const v8h*)(arow + kc + half * 16),
                          *(const v8h*)(arow + kc + half * 16 + 8));
            // B 32x16 layout: N=row, lanes 0-15 K=kc..kc+15, lanes16-31 K=kc+16..kc+31
            v16h b = cat8(*(const v8h*)(arow + kc + half * 16),
                          *(const v8h*)(arow + kc + half * 16 + 8));
            acc = __builtin_amdgcn_wmma_f32_16x16x32_f16(
                      false, a, false, b, (short)0, acc, false, false);
        }
        #pragma unroll
        for (int r = 0; r < 8; ++r)
            tokdot[(r + half * 8) * 16 + row] = acc[r];
    } else {
        const int bi = tile >> 6;
        const int n0 = (tile & 63) << 4;
        const v4f* brow =
            (const v4f*)&x[((size_t)bi * NNODE + n0 + row) * DDIM];
        for (int kc = 0; kc < DDIM; kc += 32) {
            v16h a = cat8(*(const v8h*)(arow + kc + half * 16),
                          *(const v8h*)(arow + kc + half * 16 + 8));
            const int q = (kc + half * 16) >> 2;     // v4f index of first float
            v16h b = cvt16(brow[q], brow[q + 1], brow[q + 2], brow[q + 3]);
            acc = __builtin_amdgcn_wmma_f32_16x16x32_f16(
                      false, a, false, b, (short)0, acc, false, false);
        }
        #pragma unroll
        for (int r = 0; r < 8; ++r)
            crossdot[((size_t)bi * 16 + r + half * 8) * NNODE + n0 + row] = acc[r];
    }
}

// ---------------------------------------------------------------------------
// Kernel 2: per-batch score + exact top-k (jax.lax.top_k semantics: values
// descending, stable tie-break by lower index) via O(M^2) rank counting.
// One wave per row for coalesced score dots.
// ---------------------------------------------------------------------------
__global__ __launch_bounds__(1024) void score_topk_kernel(
        const float* __restrict__ x, const float* __restrict__ tokens,
        const float* __restrict__ pool_w,
        int* __restrict__ perm, float* __restrict__ tsc)
{
    __shared__ float sw[DDIM];
    __shared__ float sc[MTOT];
    __shared__ float snorm;
    const int tid  = threadIdx.x;
    const int wave = tid >> 5;
    const int lane = tid & 31;
    const int b = blockIdx.x;

    if (tid < DDIM) sw[tid] = pool_w[tid];
    __syncthreads();
    if (tid == 0) {
        float s = 0.f;
        for (int i = 0; i < DDIM; ++i) s += sw[i] * sw[i];
        snorm = sqrtf(s);
    }
    __syncthreads();
    const float inv = 1.0f / snorm;

    // one wave per row; lane-strided (coalesced) reads + shuffle reduction
    for (int m = wave; m < MTOT; m += 32) {
        const float* rowp = (m < T_TOK)
            ? (tokens + m * DDIM)
            : (x + ((size_t)b * NNODE + (m - T_TOK)) * DDIM);
        float d = 0.f;
        #pragma unroll
        for (int i = 0; i < DDIM / 32; ++i)
            d += rowp[lane + 32 * i] * sw[lane + 32 * i];
        #pragma unroll
        for (int off = 16; off > 0; off >>= 1)
            d += __shfl_xor(d, off, 32);
        if (lane == 0) sc[m] = tanhf(d * inv);
    }
    __syncthreads();

    for (int m = tid; m < MTOT; m += blockDim.x) {
        const float s = sc[m];
        int cnt = 0;
        for (int j = 0; j < MTOT; ++j) {
            const float sj = sc[j];
            cnt += ((sj > s) || (sj == s && j < m)) ? 1 : 0;
        }
        if (cnt < KSEL) { perm[b * KSEL + cnt] = m; tsc[b * KSEL + cnt] = s; }
    }
}

// ---------------------------------------------------------------------------
// Kernel 3: xp = xf[perm] * top_scores  (one block per pooled row)
// Also writes the batch vector (third output) as float.
// ---------------------------------------------------------------------------
__global__ __launch_bounds__(256) void gather_x_kernel(
        const float* __restrict__ x, const float* __restrict__ tokens,
        const int* __restrict__ perm, const float* __restrict__ tsc,
        float* __restrict__ out)
{
    const int br = blockIdx.x;            // b*KSEL + r
    const int b = br / KSEL;
    const int p = perm[br];
    const float s = tsc[br];
    const float* src = (p < T_TOK)
        ? (tokens + p * DDIM)
        : (x + ((size_t)b * NNODE + (p - T_TOK)) * DDIM);
    out[(size_t)br * DDIM + threadIdx.x] = src[threadIdx.x] * s;
    if (threadIdx.x == 0) {
        float* out_batch = out + (size_t)BATCH * KSEL * DDIM
                               + (size_t)BATCH * KSEL * KSEL;
        out_batch[br] = (float)b;
    }
}

// ---------------------------------------------------------------------------
// Kernel 4: Ap[b,r,c] = A[b, perm[r], perm[c]]  (one block per (b,r) row)
// ---------------------------------------------------------------------------
__global__ __launch_bounds__(544) void gather_A_kernel(
        const int* __restrict__ adj, const float* __restrict__ crossdot,
        const float* __restrict__ tokdot, const int* __restrict__ perm,
        float* __restrict__ outA)
{
    const int br = blockIdx.x;            // b*KSEL + r
    const int b = br / KSEL, r = br % KSEL;
    const int c = threadIdx.x;
    if (c >= KSEL) return;
    const int pr = perm[b * KSEL + r];
    const int pc = perm[b * KSEL + c];
    float v;
    if (pr < T_TOK) {
        if (pc < T_TOK)
            v = (tokdot[pr * 16 + pc] >= INNER_TH) ? 1.f : 0.f;
        else
            v = (crossdot[((size_t)b * 16 + pr) * NNODE + (pc - T_TOK)] >= CROSS_TH)
                    ? 1.f : 0.f;
    } else {
        if (pc < T_TOK)
            v = 0.f;
        else
            v = (adj[((size_t)b * NNODE + (pr - T_TOK)) * NNODE + (pc - T_TOK)] != 0)
                    ? 1.f : 0.f;
    }
    outA[(size_t)br * KSEL + c] = v;
}

extern "C" void kernel_launch(void* const* d_in, const int* in_sizes, int n_in,
                              void* d_out, int out_size, void* d_ws, size_t ws_size,
                              hipStream_t stream)
{
    (void)in_sizes; (void)n_in; (void)out_size; (void)ws_size;
    const float* x      = (const float*)d_in[0];
    const int*   adj    = (const int*)d_in[1];
    const float* tokens = (const float*)d_in[2];
    const float* pool_w = (const float*)d_in[3];
    float* out = (float*)d_out;

    // Workspace layout (floats): crossdot [B*16*N] | tokdot [256] | perm [B*k] | tsc [B*k]
    float* crossdot = (float*)d_ws;
    float* tokdot   = crossdot + (size_t)BATCH * 16 * NNODE;
    int*   perm     = (int*)(tokdot + 256);
    float* tsc      = (float*)(perm + BATCH * KSEL);

    cross_wmma_kernel<<<BATCH * (NNODE / 16) + 1, 32, 0, stream>>>(
        x, tokens, crossdot, tokdot);
    score_topk_kernel<<<BATCH, 1024, 0, stream>>>(x, tokens, pool_w, perm, tsc);
    gather_x_kernel<<<BATCH * KSEL, 256, 0, stream>>>(x, tokens, perm, tsc, out);
    gather_A_kernel<<<BATCH * KSEL, 544, 0, stream>>>(
        adj, crossdot, tokdot, perm, out + (size_t)BATCH * KSEL * DDIM);
}
